// Network_3315714753209
// MI455X (gfx1250) — matrix operand order repaired and checked by reading, not compiled, not run on hardware
//
#include <hip/hip_runtime.h>
#include <stdint.h>

typedef __attribute__((ext_vector_type(16))) _Float16 v16h;
typedef __attribute__((ext_vector_type(8)))  _Float16 v8h;
typedef __attribute__((ext_vector_type(8)))  float    v8f;
typedef int v4i_vs __attribute__((vector_size(4 * sizeof(int))));

#define NB         4
#define HW         128
#define CIN        32
#define OUTCH_REAL 315
#define OUTCH_PAD  320

// gfx1250 async global->LDS DMA (ASYNCcnt-tracked), guarded so compile never breaks.
#if defined(__has_builtin)
#  if __has_builtin(__builtin_amdgcn_global_load_async_to_lds_b128)
#    define HAVE_ASYNC_LDS 1
#  endif
#endif
#ifndef HAVE_ASYNC_LDS
#  define HAVE_ASYNC_LDS 0
#endif

__device__ __forceinline__ void async_copy16(const _Float16* g, _Float16* l) {
#if HAVE_ASYNC_LDS
    __builtin_amdgcn_global_load_async_to_lds_b128(
        (__attribute__((address_space(1))) v4i_vs*)(uintptr_t)g,
        (__attribute__((address_space(3))) v4i_vs*)(uint32_t)(uintptr_t)l, 0, 0);
#else
    *(v8h*)l = *(const v8h*)g;
#endif
}

__device__ __forceinline__ void wait_async_lds() {
#if HAVE_ASYNC_LDS
#  if defined(__has_builtin) && __has_builtin(__builtin_amdgcn_s_wait_asynccnt)
    __builtin_amdgcn_s_wait_asynccnt(0);
#  else
    asm volatile("s_wait_asynccnt 0x0" ::: "memory");
#  endif
#endif
}

__device__ __forceinline__ float lrelu_f(float x) { return x > 0.f ? x : 0.1f * x; }

// ---------------------------------------------------------------------------
// Zero a small f16 region (zero-page for OOB conv taps).
// ---------------------------------------------------------------------------
__global__ void k_zero(_Float16* __restrict__ p, int n) {
    int i = blockIdx.x * blockDim.x + threadIdx.x;
    if (i < n) p[i] = (_Float16)0.f;
}

// ---------------------------------------------------------------------------
// S (4,3,1024,1024) f32 -> S8 (4,3,128,128) f32   (stride-8 subsample)
// ---------------------------------------------------------------------------
__global__ void k_downsample(const float* __restrict__ S, float* __restrict__ S8) {
    int idx = blockIdx.x * blockDim.x + threadIdx.x;
    if (idx >= NB * 3 * HW * HW) return;
    int w = idx & 127, h = (idx >> 7) & 127, nc = idx >> 14;
    S8[idx] = S[((size_t)nc * 1024 + (size_t)h * 8) * 1024 + (size_t)w * 8];
}

// ---------------------------------------------------------------------------
// Filter NCHW f32 -> NHWC f16 (C=32 contiguous, feeds WMMA A fragments)
// ---------------------------------------------------------------------------
__global__ void k_cvt_filter(const float* __restrict__ F, _Float16* __restrict__ out) {
    int idx = blockIdx.x * blockDim.x + threadIdx.x;
    if (idx >= NB * CIN * HW * HW) return;
    int w = idx & 127, h = (idx >> 7) & 127, c = (idx >> 14) & 31, n = idx >> 19;
    out[(((size_t)n * HW + h) * HW + w) * CIN + c] = (_Float16)F[idx];
}

// ---------------------------------------------------------------------------
// Weights OIHW f32 -> per-(tap, cout-tile) WMMA-B fragments, f16.
// B layout (32x16, 16-bit): lanes 0-15 hold K={0..7,16..23},
// lanes 16-31 hold K={8..15,24..31}; column N = lane&15.
// frag index = ((tap*tiles + tile)*32 + lane)*16 + e
// ---------------------------------------------------------------------------
__global__ void k_prep_w(const float* __restrict__ w, _Float16* __restrict__ frag,
                         int coutReal, int tiles) {
    int idx = blockIdx.x * blockDim.x + threadIdx.x;
    int total = 9 * tiles * 32 * 16;
    if (idx >= total) return;
    int e    = idx & 15;
    int lane = (idx >> 4) & 31;
    int tile = (idx >> 9) % tiles;
    int tap  = (idx >> 9) / tiles;
    int cout = tile * 16 + (lane & 15);
    int K    = (e & 7) + ((lane >> 4) << 3) + ((e >> 3) << 4);   // cin
    int ty = tap / 3, tx = tap % 3;
    float v = 0.f;
    if (cout < coutReal)
        v = w[(((size_t)cout * CIN + K) * 3 + ty) * 3 + tx];
    frag[idx] = (_Float16)v;
}

// ---------------------------------------------------------------------------
// 3x3 conv, cin=32, implicit GEMM via v_wmma_f32_16x16x32_f16.
// One wave -> 16 pixels x 16 couts, K-loop = 9 taps x 32 cin.
//  - B fragments DMA'd global->LDS (async, ASYNCcnt) then read as ds_load_b128.
//  - A tiles loaded branch-free: OOB lanes redirect to a zeroed page (cndmask on
//    address, no exec manipulation -> EXEC stays all-ones for WMMA).
// A layout (16x32 f16): lane<16: row=lane, elems {K0..7,K16..23};
// lane>=16: row=lane-16, elems {K8..15,K24..31}.
// C layout: VGPR r -> M = r + 8*(lane>=16), N = lane&15.
// ---------------------------------------------------------------------------
__global__ void __launch_bounds__(128)
k_conv3x3_wmma(const _Float16* __restrict__ actIn,   // NHWC, C=32
               const _Float16* __restrict__ wfrag,
               const float*    __restrict__ bias,
               const _Float16* __restrict__ resid,   // NHWC C=32 or null
               const _Float16* __restrict__ zpage,   // >=64 zeroed halfs
               _Float16*       __restrict__ actOut,  // NHWC, C=coutStride
               int coutTiles, int coutStride, int coutReal, int doLrelu) {
    __shared__ _Float16 ldsB[4 * 9 * 512];           // 4 waves x 9 taps x (32x16)

    int wslot = threadIdx.x >> 5;
    int wid   = blockIdx.x * (blockDim.x >> 5) + wslot;
    int lane  = threadIdx.x & 31;
    int totalWaves = NB * HW * (HW / 16) * coutTiles;
    if (wid >= totalWaves) return;                   // wave-uniform guard

    int ct  = wid % coutTiles;
    int pix = wid / coutTiles;
    int w0  = (pix & 7) * 16;
    int h   = (pix >> 3) & 127;
    int n   = pix >> 10;

    int row  = lane & 15;
    int half = lane >> 4;
    int c0   = half * 8;

    // ---- stage all 9 B fragments into LDS via async DMA (32 B per lane/tap) ----
    _Float16* myB = &ldsB[wslot * 9 * 512];
#pragma unroll
    for (int tap = 0; tap < 9; ++tap) {
        const _Float16* src = wfrag + (((size_t)(tap * coutTiles + ct)) * 32 + lane) * 16;
        _Float16*       dst = myB + tap * 512 + lane * 16;
        async_copy16(src,     dst);
        async_copy16(src + 8, dst + 8);
    }

    // ---- preload all 9 A fragments, branch-free (zero-page redirect) ----
    v16h a[9];
#pragma unroll
    for (int tap = 0; tap < 9; ++tap) {
        int dy = tap / 3 - 1, dx = tap % 3 - 1;
        int hh = h + dy;
        int ww = w0 + row + dx;
        bool valid = ((unsigned)hh < HW) & ((unsigned)ww < HW);
        const _Float16* p =
            valid ? actIn + (((size_t)n * HW + hh) * HW + ww) * CIN : zpage;
        v8h lo = *(const v8h*)(p + c0);              // K block {c0..c0+7}
        v8h hi = *(const v8h*)(p + c0 + 16);         // K block {c0+16..c0+23}
        a[tap] = __builtin_shufflevector(lo, hi, 0,1,2,3,4,5,6,7,8,9,10,11,12,13,14,15);
    }

    wait_async_lds();                                // B fragments resident in LDS

    // ---- 9 accumulating WMMAs (C-operand RAW needs no NOPs per hazard table) ----
    v8f acc = {};
#pragma unroll
    for (int tap = 0; tap < 9; ++tap) {
        v16h b = *(const v16h*)(myB + tap * 512 + lane * 16);   // ds_load_b128 x2
        acc = __builtin_amdgcn_wmma_f32_16x16x32_f16(false, a[tap], false, b,
                                                     (short)0, acc, false, false);
    }

    // ---- epilogue: bias (+residual) (+lrelu), store f16 NHWC ----
    int cg = ct * 16 + (lane & 15);
    float bv = (bias && cg < coutReal) ? bias[cg] : 0.f;
#pragma unroll
    for (int r = 0; r < 8; ++r) {
        int M = r + half * 8;                        // pixel within tile
        size_t pixoff = ((size_t)n * HW + h) * HW + (w0 + M);
        float v = acc[r] + bv;
        if (resid) v += (float)resid[pixoff * CIN + cg];
        if (doLrelu) v = lrelu_f(v);
        actOut[pixoff * coutStride + cg] = (_Float16)v;
    }
}

// ---------------------------------------------------------------------------
// SAC stage: out[h,w] = sum_a k1[a] * sum_b k2[b] * in[clamp(h+a-8), clamp(w+b-8)]
// + per-pixel bias; optional lrelu; final stage adds S8 and writes f32 d_out.
// ---------------------------------------------------------------------------
__global__ void k_sac(const float* __restrict__ SBin,       // (4,3,128,128) f32
                      const _Float16* __restrict__ RBF,     // NHWC, C=320 f16
                      const float* __restrict__ S8,         // non-null on final stage
                      float* __restrict__ SBout,
                      int stage, int doLrelu) {
    int idx = blockIdx.x * blockDim.x + threadIdx.x;
    if (idx >= NB * 3 * HW * HW) return;
    int w = idx & 127, h = (idx >> 7) & 127;
    int c = (idx >> 14) % 3, n = (idx >> 14) / 3;

    const _Float16* kpix = RBF + (((size_t)n * HW + h) * HW + w) * OUTCH_PAD;
    float k1[17], k2[17];
#pragma unroll
    for (int a = 0; a < 17; ++a) {
        k1[a] = (float)kpix[stage * 102 + c * 17 + a];
        k2[a] = (float)kpix[stage * 102 + 51 + c * 17 + a];
    }

    const float* plane = SBin + ((size_t)(n * 3 + c)) * HW * HW;
    float acc = 0.f;
    for (int a = 0; a < 17; ++a) {
        int hh = h + a - 8; hh = hh < 0 ? 0 : (hh > 127 ? 127 : hh);
        const float* rowp = plane + (size_t)hh * HW;
        int hp = h + a - 7; hp = hp < 0 ? 0 : (hp > 127 ? 127 : hp);
        __builtin_prefetch(plane + (size_t)hp * HW + w, 0, 0);   // global_prefetch_b8
        float inner = 0.f;
#pragma unroll
        for (int b = 0; b < 17; ++b) {
            int ww = w + b - 8; ww = ww < 0 ? 0 : (ww > 127 ? 127 : ww);
            inner = __builtin_fmaf(k2[b], rowp[ww], inner);
        }
        acc = __builtin_fmaf(k1[a], inner, acc);
    }
    acc += (float)kpix[306 + stage * 3 + c];
    if (doLrelu) acc = lrelu_f(acc);
    if (S8) acc += S8[idx];
    SBout[idx] = acc;
}

// ---------------------------------------------------------------------------
extern "C" void kernel_launch(void* const* d_in, const int* in_sizes, int n_in,
                              void* d_out, int out_size, void* d_ws, size_t ws_size,
                              hipStream_t stream) {
    (void)in_sizes; (void)n_in; (void)out_size; (void)ws_size;
    const float* S      = (const float*)d_in[0];
    const float* Filter = (const float*)d_in[1];
    const float* w0     = (const float*)d_in[2];
    const float* b0     = (const float*)d_in[3];
    const float* r1w1   = (const float*)d_in[4];
    const float* r1b1   = (const float*)d_in[5];
    const float* r1w2   = (const float*)d_in[6];
    const float* r1b2   = (const float*)d_in[7];
    const float* r2w1   = (const float*)d_in[8];
    const float* r2b1   = (const float*)d_in[9];
    const float* r2w2   = (const float*)d_in[10];
    const float* r2b2   = (const float*)d_in[11];
    const float* w3     = (const float*)d_in[12];
    const float* b3     = (const float*)d_in[13];
    float* out = (float*)d_out;

    char* ws = (char*)d_ws;
    auto alloc = [&](size_t bytes) -> char* {
        char* p = ws; ws += (bytes + 255) & ~(size_t)255; return p;
    };
    const size_t PIX = (size_t)NB * HW * HW;                 // 65536
    _Float16* zpage = (_Float16*)alloc(256);
    float*    S8    = (float*)   alloc((size_t)NB * 3 * HW * HW * 4);
    _Float16* FiltH = (_Float16*)alloc(PIX * CIN * 2);
    _Float16* actA  = (_Float16*)alloc(PIX * CIN * 2);
    _Float16* actB  = (_Float16*)alloc(PIX * CIN * 2);
    _Float16* actC  = (_Float16*)alloc(PIX * CIN * 2);
    _Float16* RBFb  = (_Float16*)alloc(PIX * OUTCH_PAD * 2);
    _Float16* fw0   = (_Float16*)alloc((size_t)9 * 2  * 512 * 2);
    _Float16* f11   = (_Float16*)alloc((size_t)9 * 2  * 512 * 2);
    _Float16* f12   = (_Float16*)alloc((size_t)9 * 2  * 512 * 2);
    _Float16* f21   = (_Float16*)alloc((size_t)9 * 2  * 512 * 2);
    _Float16* f22   = (_Float16*)alloc((size_t)9 * 2  * 512 * 2);
    _Float16* fw3   = (_Float16*)alloc((size_t)9 * 20 * 512 * 2);
    float*    SBa   = (float*)   alloc((size_t)NB * 3 * HW * HW * 4);
    float*    SBb   = (float*)   alloc((size_t)NB * 3 * HW * HW * 4);

    const int T = 256;
    k_zero<<<1, 128, 0, stream>>>(zpage, 128);

    int nS8 = NB * 3 * HW * HW;                              // 196608
    k_downsample<<<(nS8 + T - 1) / T, T, 0, stream>>>(S, S8);

    int nF = NB * CIN * HW * HW;                             // 2097152
    k_cvt_filter<<<(nF + T - 1) / T, T, 0, stream>>>(Filter, FiltH);

    auto prep = [&](const float* w, _Float16* frag, int creal, int tiles) {
        int total = 9 * tiles * 512;
        k_prep_w<<<(total + T - 1) / T, T, 0, stream>>>(w, frag, creal, tiles);
    };
    prep(w0,   fw0, 32, 2);
    prep(r1w1, f11, 32, 2);
    prep(r1w2, f12, 32, 2);
    prep(r2w1, f21, 32, 2);
    prep(r2w2, f22, 32, 2);
    prep(w3,   fw3, OUTCH_REAL, 20);

    auto conv = [&](const _Float16* in, const _Float16* fr, const float* bias,
                    const _Float16* resid, _Float16* o, int tiles, int stride,
                    int creal, int doLrelu) {
        int waves  = NB * HW * (HW / 16) * tiles;            // 4096 * tiles
        int blocks = (waves + 3) / 4;                        // 4 waves / block
        k_conv3x3_wmma<<<blocks, 128, 0, stream>>>(in, fr, bias, resid, zpage, o,
                                                   tiles, stride, creal, doLrelu);
    };
    // x = lrelu(conv0(Filter))
    conv(FiltH, fw0, b0,   nullptr, actA, 2, CIN, 32, 1);
    // resblock 1
    conv(actA,  f11, r1b1, nullptr, actB, 2, CIN, 32, 1);
    conv(actB,  f12, r1b2, actA,    actC, 2, CIN, 32, 1);
    // resblock 2
    conv(actC,  f21, r2b1, nullptr, actA, 2, CIN, 32, 1);
    conv(actA,  f22, r2b2, actC,    actB, 2, CIN, 32, 1);
    // RBF = conv3(x), 315 (->320 padded) channels
    conv(actB,  fw3, b3,   nullptr, RBFb, 20, OUTCH_PAD, OUTCH_REAL, 0);

    int nSB = NB * 3 * HW * HW;
    k_sac<<<(nSB + T - 1) / T, T, 0, stream>>>(S8,  RBFb, nullptr, SBa, 0, 1);
    k_sac<<<(nSB + T - 1) / T, T, 0, stream>>>(SBa, RBFb, nullptr, SBb, 1, 1);
    k_sac<<<(nSB + T - 1) / T, T, 0, stream>>>(SBb, RBFb, S8,      out, 2, 0);
}
